// TwoStageGatedMoE_37099927503251
// MI455X (gfx1250) — matrix operand order repaired and checked by reading, not compiled, not run on hardware
//
#include <hip/hip_runtime.h>
#include <math.h>

// ---------------- problem constants ----------------
#define NB   32
#define LL   128
#define TT   4096      // N*L tokens
#define DD   256
#define HH   1024
#define GG   4
#define EGU  4
#define EE   16        // G*EG experts
#define K1S  2

// d_out layout: final (TT*DD) | lb1 | lb2 | gate1_usage(4) | gate2_usage(16)
#define OUT_FINAL (TT * DD)

// ws layout: comb (TT*EE floats) | acc (48 floats)
#define WS_COMB_FLOATS (TT * EE)
// acc layout: [0]=nvalid [1..4]=p1sum [5..8]=cnt(sel) [9..24]=p2sum [25..40]=usage2

typedef __bf16 bf16_t;
typedef __attribute__((ext_vector_type(16))) __bf16      v16bf;
typedef __attribute__((ext_vector_type(8)))  float       v8f;
typedef __attribute__((ext_vector_type(4)))  unsigned int u32x4;

struct B256 { u32x4 lo; u32x4 hi; };

static __device__ inline v16bf ld_frag(const unsigned short* p0,
                                       const unsigned short* p1) {
  B256 b;
  b.lo = *(const u32x4*)p0;
  b.hi = *(const u32x4*)p1;
  return __builtin_bit_cast(v16bf, b);
}

static __device__ inline unsigned short f2bf(float f) {
  unsigned int u = __float_as_uint(f);
  u += 0x7FFFu + ((u >> 16) & 1u);   // round-to-nearest-even
  return (unsigned short)(u >> 16);
}

static __device__ inline float gelu_exact(float v) {
  return 0.5f * v * (1.0f + erff(v * 0.7071067811865476f));
}

static __device__ inline v8f wmma_bf16(v16bf a, v16bf b, v8f c) {
  return __builtin_amdgcn_wmma_f32_16x16x32_bf16(false, a, false, b,
                                                 (short)0, c, false, false);
}

// ---------------- kernel 0: zero accumulators ----------------
__global__ void init_acc_kernel(float* __restrict__ acc) {
  int i = threadIdx.x;
  if (i < 48) acc[i] = 0.0f;
}

// ---------------- kernel 1: gating + aux-loss stats ----------------
__global__ __launch_bounds__(256) void gate_kernel(
    const float* __restrict__ x, const float* __restrict__ adj,
    const float* __restrict__ tf, const float* __restrict__ us,
    const float* __restrict__ mask,
    const float* __restrict__ g1W, const float* __restrict__ g1b,
    const float* __restrict__ g2W, const float* __restrict__ g2b,
    float* __restrict__ comb, float* __restrict__ acc) {
  __shared__ float4 s1w[515];    // gate1_W (515,4) as float4 rows
  __shared__ float4 s2w[2048];   // gate2_W (4,512,4) as float4 rows
  __shared__ float  s1b[4];
  __shared__ float  s2b[16];
  __shared__ float  red[48];

  const int tid = threadIdx.x;
  const float4* g1w4 = (const float4*)g1W;
  const float4* g2w4 = (const float4*)g2W;
  for (int i = tid; i < 515; i += 256) s1w[i] = g1w4[i];
  for (int i = tid; i < 2048; i += 256) s2w[i] = g2w4[i];
  if (tid < 4)  s1b[tid] = g1b[tid];
  if (tid < 16) s2b[tid] = g2b[tid];
  if (tid < 48) red[tid] = 0.0f;
  __syncthreads();

  const int t = blockIdx.x * 256 + tid;   // 16 blocks -> exactly TT tokens
  const float* xp = x  + (long)t * DD;
  const float* tp = tf + (long)t * DD;
  const float* up = us + (long)t * DD;

  float z1[4] = {0.f, 0.f, 0.f, 0.f};
  float z2[4][4];
  #pragma unroll
  for (int g = 0; g < 4; ++g)
    #pragma unroll
    for (int j = 0; j < 4; ++j) z2[g][j] = 0.f;

  for (int d = 0; d < DD; ++d) {
    float xv = xp[d];
    float4 w = s1w[d];
    z1[0] += xv * w.x; z1[1] += xv * w.y; z1[2] += xv * w.z; z1[3] += xv * w.w;
    #pragma unroll
    for (int g = 0; g < 4; ++g) {
      float4 w2 = s2w[g * 512 + d];
      z2[g][0] += xv * w2.x; z2[g][1] += xv * w2.y;
      z2[g][2] += xv * w2.z; z2[g][3] += xv * w2.w;
    }
  }
  for (int j = 0; j < 3; ++j) {
    float av = adj[t * 3 + j];
    float4 w = s1w[256 + j];
    z1[0] += av * w.x; z1[1] += av * w.y; z1[2] += av * w.z; z1[3] += av * w.w;
  }
  for (int d = 0; d < DD; ++d) {
    float tv = tp[d];
    float4 w = s1w[259 + d];
    z1[0] += tv * w.x; z1[1] += tv * w.y; z1[2] += tv * w.z; z1[3] += tv * w.w;
  }
  for (int d = 0; d < DD; ++d) {
    float uv = up[d];
    #pragma unroll
    for (int g = 0; g < 4; ++g) {
      float4 w2 = s2w[g * 512 + 256 + d];
      z2[g][0] += uv * w2.x; z2[g][1] += uv * w2.y;
      z2[g][2] += uv * w2.z; z2[g][3] += uv * w2.w;
    }
  }

  // stage-1 softmax over G=4
  float p1[4];
  {
    float mx = z1[0] + s1b[0];
    float zz[4];
    #pragma unroll
    for (int g = 0; g < 4; ++g) { zz[g] = z1[g] + s1b[g]; mx = fmaxf(mx, zz[g]); }
    float s = 0.f;
    #pragma unroll
    for (int g = 0; g < 4; ++g) { p1[g] = expf(zz[g] - mx); s += p1[g]; }
    float inv = 1.0f / s;
    #pragma unroll
    for (int g = 0; g < 4; ++g) p1[g] *= inv;
  }
  // top-2 (first-occurrence tie break, matches jax top_k)
  int i1 = 0;
  for (int g = 1; g < 4; ++g) if (p1[g] > p1[i1]) i1 = g;
  int i2 = -1;
  for (int g = 0; g < 4; ++g) {
    if (g == i1) continue;
    if (i2 < 0 || p1[g] > p1[i2]) i2 = g;
  }

  // stage-2 softmax / argmax per group
  float p2[4][4], val2[4];
  int idx2[4];
  #pragma unroll
  for (int g = 0; g < 4; ++g) {
    float zz[4];
    float mx = -3.4e38f;
    #pragma unroll
    for (int j = 0; j < 4; ++j) { zz[j] = z2[g][j] + s2b[g * 4 + j]; mx = fmaxf(mx, zz[j]); }
    float s = 0.f;
    #pragma unroll
    for (int j = 0; j < 4; ++j) { p2[g][j] = expf(zz[j] - mx); s += p2[g][j]; }
    float inv = 1.0f / s;
    #pragma unroll
    for (int j = 0; j < 4; ++j) p2[g][j] *= inv;
    int bi = 0;
    for (int j = 1; j < 4; ++j) if (p2[g][j] > p2[g][bi]) bi = j;
    idx2[g] = bi;
    val2[g] = p2[g][bi];
  }

  // combine weights per (token, expert)
  float cv[16];
  #pragma unroll
  for (int e = 0; e < 16; ++e) cv[e] = 0.f;
  cv[i1 * 4 + idx2[i1]] = p1[i1] * val2[i1];
  cv[i2 * 4 + idx2[i2]] = p1[i2] * val2[i2];
  float* cp = comb + (long)t * EE;
  #pragma unroll
  for (int e = 0; e < 16; ++e) cp[e] = cv[e];

  // aux-loss statistics
  float vf = (mask[t] > 0.0f) ? 1.0f : 0.0f;
  atomicAdd(&red[0], vf);
  #pragma unroll
  for (int g = 0; g < 4; ++g) atomicAdd(&red[1 + g], p1[g] * vf);
  int sels[2] = {i1, i2};
  #pragma unroll
  for (int k = 0; k < 2; ++k) {
    int s = sels[k];
    atomicAdd(&red[5 + s], vf);                          // cnt / gate1_usage
    #pragma unroll
    for (int j = 0; j < 4; ++j) atomicAdd(&red[9 + s * 4 + j], p2[s][j] * vf);
    atomicAdd(&red[25 + s * 4 + idx2[s]], vf);           // gate2_usage
  }
  __syncthreads();
  if (tid < 48) atomicAdd(&acc[tid], red[tid]);
}

// ---------------- kernel 2: finalize aux losses ----------------
__global__ void finalize_kernel(const float* __restrict__ acc,
                                float* __restrict__ out) {
  if (threadIdx.x != 0 || blockIdx.x != 0) return;
  float nv = fmaxf(acc[0], 1.0f);
  float lb1 = 0.f;
  for (int g = 0; g < 4; ++g) {
    float pm = acc[1 + g] / nv - 0.25f;
    lb1 += pm * pm;
  }
  float lb2 = 0.f;
  for (int g = 0; g < 4; ++g) {
    float c = acc[5 + g];
    if (c > 0.0f) {
      float dn = fmaxf(c, 1.0f);
      float s = 0.f;
      for (int j = 0; j < 4; ++j) {
        float pm = acc[9 + g * 4 + j] / dn - 0.25f;
        s += pm * pm;
      }
      lb2 += s;
    }
  }
  out[OUT_FINAL + 0] = lb1;
  out[OUT_FINAL + 1] = lb2;
  for (int g = 0; g < 4; ++g)  out[OUT_FINAL + 2 + g] = acc[5 + g];
  for (int k = 0; k < 16; ++k) out[OUT_FINAL + 6 + k] = acc[25 + k];
}

// ---------------- kernel 3: fused expert FFN (bf16 WMMA) ----------------
// block = 128 threads (4 wave32), 16 tokens per block, 256 blocks.
// Per expert: loop 16 H-chunks of 64: h = gelu(x @ W1chunk + b1) via WMMA,
// then partial out += h @ W2chunk; finally final += comb * (out + b2).
#define XPITCH 264
#define HPITCH 72
#define W1PITCH 264   // w1t[c][d], c in [0,64), d in [0,256)
#define W2PITCH 72    // w2t[n][k], n in [0,256), k in [0,64)

__global__ __launch_bounds__(128) void moe_ffn_kernel(
    const float* __restrict__ x, const float* __restrict__ comb,
    const float* __restrict__ W1, const float* __restrict__ b1,
    const float* __restrict__ W2, const float* __restrict__ b2,
    float* __restrict__ out) {
  __shared__ alignas(16) unsigned short xs[16 * XPITCH];   //  8448 B
  __shared__ alignas(16) unsigned short hs[16 * HPITCH];   //  2304 B
  __shared__ alignas(16) unsigned short wb[18432];         // 36864 B (shared W1T/W2T)
  __shared__ float combs[16 * 16];                          //  1024 B
  __shared__ int   act[16];

  const int tid  = threadIdx.x;
  const int wv   = tid >> 5;
  const int lane = tid & 31;
  const int lo   = lane & 15;
  const int hi   = lane >> 4;
  const int t0   = blockIdx.x * 16;

  // load x tile (fp32 -> bf16) and combine weights
  for (int i = tid; i < 16 * 256; i += 128) {
    int m = i >> 8, d = i & 255;
    xs[m * XPITCH + d] = f2bf(x[(long)(t0 + m) * DD + d]);
  }
  for (int i = tid; i < 256; i += 128) combs[i] = comb[(long)t0 * EE + i];
  __syncthreads();
  if (tid < 16) {
    int a = 0;
    for (int m = 0; m < 16; ++m) a |= (combs[m * 16 + tid] != 0.0f) ? 1 : 0;
    act[tid] = a;
  }
  __syncthreads();

  const v8f vzero = {0.f, 0.f, 0.f, 0.f, 0.f, 0.f, 0.f, 0.f};
  v8f facc[4];
  #pragma unroll
  for (int nt = 0; nt < 4; ++nt) facc[nt] = vzero;

  for (int e = 0; e < EE; ++e) {
    if (!act[e]) continue;

    v8f a2[4];
    #pragma unroll
    for (int nt = 0; nt < 4; ++nt) a2[nt] = vzero;

    const float* w1e = W1 + (long)e * DD * HH;
    const float* w2e = W2 + (long)e * HH * DD;

    for (int hc = 0; hc < HH / 64; ++hc) {
      __syncthreads();   // protect wb & hs reuse from previous phase
      // ---- load W1 chunk transposed: wb[c*264 + d] = W1[e][d][hc*64 + c]
      {
        const float* w1p = w1e + hc * 64;
        for (int i = tid; i < 64 * 256; i += 128) {
          int d = i >> 6, c = i & 63;
          wb[c * W1PITCH + d] = f2bf(w1p[(long)d * HH + c]);
        }
      }
      __syncthreads();
      // ---- stage 1: wave wv computes h tile, cols [wv*16, wv*16+16)
      {
        const int cb = wv * 16;
        const unsigned short* arow = xs + lo * XPITCH;
        const unsigned short* brow = wb + (cb + lo) * W1PITCH;
        v8f acc = vzero;
        #pragma unroll
        for (int k = 0; k < 256; k += 32) {
          v16bf af = ld_frag(arow + k + hi * 8, arow + k + 16 + hi * 8);
          v16bf bf = ld_frag(brow + k + hi * 8, brow + k + 16 + hi * 8);
          acc = wmma_bf16(af, bf, acc);
        }
        const int ccol = cb + lo;
        float b1v = b1[e * HH + hc * 64 + ccol];
        #pragma unroll
        for (int r = 0; r < 8; ++r) {
          int m = hi * 8 + r;
          hs[m * HPITCH + ccol] = f2bf(gelu_exact(acc[r] + b1v));
        }
      }
      __syncthreads();
      // ---- load W2 chunk transposed: wb[n*72 + k] = W2[e][hc*64 + k][n]
      {
        const float* w2p = w2e + (long)hc * 64 * DD;
        for (int i = tid; i < 64 * 256; i += 128) {
          int k = i >> 8, n = i & 255;
          wb[n * W2PITCH + k] = f2bf(w2p[(long)k * DD + n]);
        }
      }
      __syncthreads();
      // ---- stage 2 partial accumulate: wave wv owns 4 output tiles
      {
        const unsigned short* arow = hs + lo * HPITCH;
        #pragma unroll
        for (int kk = 0; kk < 64; kk += 32) {
          v16bf af = ld_frag(arow + kk + hi * 8, arow + kk + 16 + hi * 8);
          #pragma unroll
          for (int nt = 0; nt < 4; ++nt) {
            const unsigned short* brow = wb + ((wv * 4 + nt) * 16 + lo) * W2PITCH;
            v16bf bf = ld_frag(brow + kk + hi * 8, brow + kk + 16 + hi * 8);
            a2[nt] = wmma_bf16(af, bf, a2[nt]);
          }
        }
      }
    }

    // ---- weighted combine: final += comb[t][e] * (out_e + b2)
    #pragma unroll
    for (int nt = 0; nt < 4; ++nt) {
      int dcol = (wv * 4 + nt) * 16 + lo;
      float b2v = b2[e * DD + dcol];
      #pragma unroll
      for (int r = 0; r < 8; ++r) {
        int m = hi * 8 + r;
        float cw = combs[m * 16 + e];
        facc[nt][r] += cw * (a2[nt][r] + b2v);
      }
    }
  }

  // ---- write final
  #pragma unroll
  for (int nt = 0; nt < 4; ++nt) {
    int dcol = (wv * 4 + nt) * 16 + lo;
    #pragma unroll
    for (int r = 0; r < 8; ++r) {
      int m = hi * 8 + r;
      out[(long)(t0 + m) * DD + dcol] = facc[nt][r];
    }
  }
}

// ---------------- host launch ----------------
extern "C" void kernel_launch(void* const* d_in, const int* in_sizes, int n_in,
                              void* d_out, int out_size, void* d_ws, size_t ws_size,
                              hipStream_t stream) {
  (void)in_sizes; (void)n_in; (void)out_size; (void)ws_size;
  const float* x    = (const float*)d_in[0];
  const float* adj  = (const float*)d_in[1];
  const float* tf   = (const float*)d_in[2];
  const float* us   = (const float*)d_in[3];
  const float* mask = (const float*)d_in[4];
  const float* g1W  = (const float*)d_in[5];
  const float* g1b  = (const float*)d_in[6];
  const float* g2W  = (const float*)d_in[7];
  const float* g2b  = (const float*)d_in[8];
  const float* W1   = (const float*)d_in[9];
  const float* b1   = (const float*)d_in[10];
  const float* W2   = (const float*)d_in[11];
  const float* b2   = (const float*)d_in[12];
  float* out  = (float*)d_out;
  float* comb = (float*)d_ws;
  float* acc  = comb + WS_COMB_FLOATS;

  init_acc_kernel<<<1, 64, 0, stream>>>(acc);
  gate_kernel<<<TT / 256, 256, 0, stream>>>(x, adj, tf, us, mask,
                                            g1W, g1b, g2W, g2b, comb, acc);
  finalize_kernel<<<1, 32, 0, stream>>>(acc, out);
  moe_ffn_kernel<<<TT / 16, 128, 0, stream>>>(x, comb, W1, b1, W2, b2, out);
}
